// CausalLM_27169963114992
// MI455X (gfx1250) — matrix operand order repaired and checked by reading, not compiled, not run on hardware
//
#include <hip/hip_runtime.h>

// Problem constants (from the reference)
#define B_ 2
#define H_ 16
#define S_ 2048
#define D_ 128

#define KBLK 32               // keys per iteration
#define WAVES 4               // waves per workgroup
#define QT 16                 // q rows per wave
#define KPITCH (D_ + 8)       // bf16 pitch for K rows (272B: 16B-aligned, conflict-free)
#define VPITCH (KBLK + 8)     // bf16 pitch for V^T rows (80B: 16B-aligned, conflict-free)

typedef __attribute__((ext_vector_type(16))) __bf16 v16bf;
typedef __attribute__((ext_vector_type(8)))  float  v8f;

#define NEG_INF (-3.3895314e38f)   // ~ finfo(fp32).min (reference mask_value)

#define WMMA_BF16(A, Bm, C) \
    __builtin_amdgcn_wmma_f32_16x16x32_bf16(false, (A), false, (Bm), (short)0, (C), false, false)

__global__ __launch_bounds__(128, 1)
void fa_causal_ctxmask_kernel(const float* __restrict__ q_g,
                              const float* __restrict__ k_g,
                              const float* __restrict__ v_g,
                              const float* __restrict__ cm_g,
                              float* __restrict__ o_g) {
    __shared__ __align__(16) __bf16 Ks[KBLK][KPITCH];     // K block (row-major, bf16)
    __shared__ __align__(16) __bf16 Vst[D_][VPITCH];      // (V*mask)^T block (d-major, bf16)
    __shared__ __align__(16) __bf16 Ps[WAVES][QT][KBLK];  // per-wave P^T staging (q-major)

    const int t    = threadIdx.x;
    const int lane = t & 31;
    const int wave = t >> 5;
    const int hi   = lane >> 4;       // wave half
    const int l15  = lane & 15;

    const int b    = blockIdx.z;
    const int h    = blockIdx.y;
    const int q0wg = blockIdx.x * (WAVES * QT);
    const int q0   = q0wg + wave * QT;
    const int qrow = q0 + l15;        // this lane's q row (N axis of all C tiles)

    const size_t bh    = (size_t)b * H_ + h;
    const float* Kbase = k_g + bh * S_ * D_;
    const float* Vbase = v_g + bh * S_ * D_;
    const float* cmb   = cm_g + (size_t)b * S_;

    // ---- Q as 4 bf16 B-fragments (B-layout: lane = N = q row; elem e -> d = j*32 + hi*16 + e) ----
    v16bf bq[4];
    {
        const float* Qp = q_g + (bh * S_ + qrow) * D_;
#pragma unroll
        for (int j = 0; j < 4; ++j)
#pragma unroll
            for (int e = 0; e < 16; e += 4) {
                const float4 f = *(const float4*)(Qp + j * 32 + hi * 16 + e);
                bq[j][e]     = (__bf16)f.x; bq[j][e + 1] = (__bf16)f.y;
                bq[j][e + 2] = (__bf16)f.z; bq[j][e + 3] = (__bf16)f.w;
            }
    }

    // ---- per-lane online softmax state (one q row per lane) + O^T accumulators ----
    float m_l = NEG_INF, l_l = 0.0f;
    v8f acc[8];                        // O^T: M = d = dt*16 + hi*8 + i, N = q row
#pragma unroll
    for (int dt = 0; dt < 8; ++dt) acc[dt] = (v8f){};

    const int nblocks = blockIdx.x * 2 + 2;   // causal: keys up to q0wg+63

    // ---- prefetch block 0 into registers ----
    const int prow = (t >> 5);         // +it*4
    const int pcol = (t & 31) * 4;
    float4 kpre[8], vpre[8];
    float  cmpre[8];
#pragma unroll
    for (int it = 0; it < 8; ++it) {
        const int row = it * 4 + prow;
        kpre[it]  = *(const float4*)(Kbase + (size_t)row * D_ + pcol);
        vpre[it]  = *(const float4*)(Vbase + (size_t)row * D_ + pcol);
        cmpre[it] = cmb[row];
    }

    for (int kb = 0; kb < nblocks; ++kb) {
        const int kb0 = kb * KBLK;

        // ---- commit prefetched block to LDS (fp32 -> bf16; fold ctx_mask into V) ----
#pragma unroll
        for (int it = 0; it < 8; ++it) {
            const int row = it * 4 + prow;
            const float4 kf = kpre[it];
            const float4 vf = vpre[it];
            const float  cv = cmpre[it];
            Ks[row][pcol + 0] = (__bf16)kf.x; Ks[row][pcol + 1] = (__bf16)kf.y;
            Ks[row][pcol + 2] = (__bf16)kf.z; Ks[row][pcol + 3] = (__bf16)kf.w;
            Vst[pcol + 0][row] = (__bf16)(vf.x * cv);
            Vst[pcol + 1][row] = (__bf16)(vf.y * cv);
            Vst[pcol + 2][row] = (__bf16)(vf.z * cv);
            Vst[pcol + 3][row] = (__bf16)(vf.w * cv);
        }
        __syncthreads();

        // ---- prefetch next block (overlaps with WMMA/softmax below) ----
        if (kb + 1 < nblocks) {
            const int nb0 = kb0 + KBLK;
#pragma unroll
            for (int it = 0; it < 8; ++it) {
                const int row = nb0 + it * 4 + prow;
                kpre[it]  = *(const float4*)(Kbase + (size_t)row * D_ + pcol);
                vpre[it]  = *(const float4*)(Vbase + (size_t)row * D_ + pcol);
                cmpre[it] = cmb[row];
            }
        }

        // ---- wave-uniform block classification (s_cmp/s_cbranch, no divergence) ----
        const bool dead = kb0 > q0 + (QT - 1);          // entire block above diagonal
        if (!dead) {
            const bool interior = (kb0 + KBLK - 1) <= q0;  // no masking needed

            // ---- S^T tiles: c0 = keys 0-15, c1 = keys 16-31 (M = key, N = q row) ----
            v8f c0 = (v8f){}, c1 = (v8f){};
#pragma unroll
            for (int j = 0; j < 4; ++j) {
                v16bf ak0, ak1;  // A-layout: lane = M = key; elem e -> d = j*32 + hi*16 + e
#pragma unroll
                for (int e = 0; e < 16; ++e) {
                    const int d = j * 32 + e + hi * 16;
                    ak0[e] = Ks[l15][d];
                    ak1[e] = Ks[l15 + 16][d];
                }
                c0 = WMMA_BF16(ak0, bq[j], c0);
                c1 = WMMA_BF16(ak1, bq[j], c1);
            }

            // ---- causal mask only on diagonal blocks ----
            float s0[8], s1[8];
            if (interior) {
#pragma unroll
                for (int i = 0; i < 8; ++i) { s0[i] = c0[i]; s1[i] = c1[i]; }
            } else {
#pragma unroll
                for (int i = 0; i < 8; ++i) {
                    s0[i] = (kb0 + hi * 8 + i      <= qrow) ? c0[i] : NEG_INF;
                    s1[i] = (kb0 + 16 + hi * 8 + i <= qrow) ? c1[i] : NEG_INF;
                }
            }

            // ---- per-lane softmax (keys in registers; one shfl per reduction) ----
            float mx = fmaxf(s0[0], s1[0]);
#pragma unroll
            for (int i = 1; i < 8; ++i) mx = fmaxf(mx, fmaxf(s0[i], s1[i]));
            mx = fmaxf(mx, __shfl_xor(mx, 16, 32));    // combine the two key-halves

            const float mnew  = fmaxf(m_l, mx);
            const float alpha = __expf(m_l - mnew);
            m_l = mnew;

            union { __bf16 hv[8]; uint4 u; } pk0, pk1;
            float rs = 0.0f;
#pragma unroll
            for (int i = 0; i < 8; ++i) {
                const float p0 = __expf(s0[i] - mnew);   // NEG_INF -> 0
                const float p1 = __expf(s1[i] - mnew);
                rs += p0 + p1;
                pk0.hv[i] = (__bf16)p0;
                pk1.hv[i] = (__bf16)p1;
            }
            rs += __shfl_xor(rs, 16, 32);
            l_l = l_l * alpha + rs;

            // P^T -> per-wave LDS (Ps[q][key]), two aligned 16B stores per lane
            *(uint4*)&Ps[wave][l15][hi * 8]      = pk0.u;
            *(uint4*)&Ps[wave][l15][16 + hi * 8] = pk1.u;

#pragma unroll
            for (int dt = 0; dt < 8; ++dt) acc[dt] *= alpha;   // per-lane scalar rescale

            // wave-private LDS RAW: DS is in-order per wave; drain DScnt (no WG barrier)
            asm volatile("s_wait_dscnt 0" ::: "memory");

            // ---- O^T += (V*mask)^T x P^T : 8 WMMAs ----
            v16bf bp;   // B-layout: lane = N = q row; elem e -> key = e + hi*16
#pragma unroll
            for (int e = 0; e < 16; ++e) bp[e] = Ps[wave][l15][e + hi * 16];

#pragma unroll
            for (int dt = 0; dt < 8; ++dt) {
                v16bf av;  // A-layout: lane = M = d-in-subtile; elem e -> key = e + hi*16
#pragma unroll
                for (int e = 0; e < 16; ++e) {
                    const int key = e + hi * 16;
                    av[e] = Vst[dt * 16 + l15][key];
                }
                acc[dt] = WMMA_BF16(av, bp, acc[dt]);
            }
        }

        __syncthreads();   // protect Ks/Vst before next commit overwrites them
    }

    // ---- epilogue: divide by softmax denominator, fp32 vector stores ----
    const float invl = 1.0f / l_l;
    float* Op = o_g + (bh * S_ + qrow) * D_ + hi * 8;
#pragma unroll
    for (int dt = 0; dt < 8; ++dt) {
        float4 r0 = { acc[dt][0] * invl, acc[dt][1] * invl, acc[dt][2] * invl, acc[dt][3] * invl };
        float4 r1 = { acc[dt][4] * invl, acc[dt][5] * invl, acc[dt][6] * invl, acc[dt][7] * invl };
        *(float4*)(Op + dt * 16)     = r0;
        *(float4*)(Op + dt * 16 + 4) = r1;
    }
}

extern "C" void kernel_launch(void* const* d_in, const int* in_sizes, int n_in,
                              void* d_out, int out_size, void* d_ws, size_t ws_size,
                              hipStream_t stream) {
    (void)in_sizes; (void)n_in; (void)out_size; (void)d_ws; (void)ws_size;
    const float* q  = (const float*)d_in[0];
    const float* k  = (const float*)d_in[1];
    const float* v  = (const float*)d_in[2];
    const float* cm = (const float*)d_in[3];
    float* o = (float*)d_out;

    dim3 grid(S_ / (WAVES * QT), H_, B_);   // (32, 16, 2)
    fa_causal_ctxmask_kernel<<<grid, WAVES * 32, 0, stream>>>(q, k, v, cm, o);
}
